// SelfAttention_48473000902942
// MI455X (gfx1250) — matrix operand order repaired and checked by reading, not compiled, not run on hardware
//
#include <hip/hip_runtime.h>
#include <hip/hip_bf16.h>

#define BB 4
#define SS 2048
#define EE 1024

typedef __attribute__((ext_vector_type(16))) __bf16 v16bf;
typedef __attribute__((ext_vector_type(8)))  __bf16 v8bf;
typedef __attribute__((ext_vector_type(8)))  float  v8f;
typedef __attribute__((ext_vector_type(4)))  float  v4f;

static __device__ __forceinline__ v16bf cat16(v8bf lo, v8bf hi) {
  return __builtin_shufflevector(lo, hi, 0,1,2,3,4,5,6,7,8,9,10,11,12,13,14,15);
}

static __device__ __forceinline__ v8f wmma_bf16(v16bf a, v16bf b, v8f c) {
  // D = A(16x32 bf16) x B(32x16 bf16) + C(16x16 f32)
  return __builtin_amdgcn_wmma_f32_16x16x32_bf16(false, a, false, b, (short)0, c,
                                                 false, false);
}

// One-lane 16B async copy global->LDS (ASYNCcnt-tracked; ISA 08_async_tensor §4).
static __device__ __forceinline__ void async_g2l_b128(unsigned lds_addr,
                                                      const void* gaddr) {
  unsigned long long ga = (unsigned long long)(size_t)gaddr;
  asm volatile("global_load_async_to_lds_b128 %0, %1, off"
               :: "v"(lds_addr), "v"(ga) : "memory");
}
static __device__ __forceinline__ void wait_async_le1() {
  asm volatile("s_wait_asynccnt 0x1" ::: "memory");
}
static __device__ __forceinline__ void wait_async_0() {
  asm volatile("s_wait_asynccnt 0x0" ::: "memory");
}

// ---------------------------------------------------------------------------
// Weight prep: Wt[n][k] = (bf16) W[k][n]
// ---------------------------------------------------------------------------
__global__ __launch_bounds__(256) void wt_prep_kernel(const float* __restrict__ W,
                                                      __bf16* __restrict__ Wt) {
  const int idx = blockIdx.x * 256 + threadIdx.x;
  const int n  = idx >> 10;
  const int kk = idx & (EE - 1);
  Wt[idx] = (__bf16)W[((size_t)kk << 10) + n];
}

// ---------------------------------------------------------------------------
// Projection: Out = X(fp32)[8192,1024] @ W[1024,1024] -> bf16.
// 8 waves, wave = 16(M)x64(N) tile. The 32x64 weight tile is shared by all
// waves: async-staged into LDS (double buffered) instead of 8x-redundant
// global loads. TRANSPOSE=1 stores V as VT[b][e][s].
// ---------------------------------------------------------------------------
template <bool TRANSPOSE>
__global__ __launch_bounds__(256) void proj_kernel(const float* __restrict__ X,
                                                   const __bf16* __restrict__ Wt,
                                                   __bf16* __restrict__ Out) {
  __shared__ __bf16 sB[2][64][40];   // [buf][n][k], row padded 32->40 (80B, 16B-aligned)

  const int tid  = threadIdx.x;
  const int lane = tid & 31;
  const int wv   = tid >> 5;
  const int l16  = lane & 15;
  const int hi   = lane >> 4;
  const int row0 = blockIdx.x * 128 + wv * 16;
  const int n0   = blockIdx.y * 64;

  // staging map: thread -> (N row 0..63, 16B chunk 0..3) of the 32x64 tile
  const int cr = tid >> 2;
  const int cc = tid & 3;

  v8f acc[4];
#pragma unroll
  for (int ct = 0; ct < 4; ++ct) acc[ct] = (v8f){0,0,0,0,0,0,0,0};

  const float* ap0 = X + (size_t)(row0 + l16) * EE;

  // preload k-chunk 0
  async_g2l_b128((unsigned)(size_t)&sB[0][cr][cc * 8],
                 Wt + (size_t)(n0 + cr) * EE + cc * 8);

  for (int kc = 0; kc < EE / 32; ++kc) {
    if (kc + 1 < EE / 32) {
      async_g2l_b128((unsigned)(size_t)&sB[(kc + 1) & 1][cr][cc * 8],
                     Wt + (size_t)(n0 + cr) * EE + (kc + 1) * 32 + cc * 8);
      wait_async_le1();              // current buffer's copy has landed
    } else {
      wait_async_0();
    }
    __syncthreads();                 // whole tile visible to all waves

    // A fragment straight from fp32 X with in-register cvt to bf16
    const int ka = kc * 32 + hi * 8;
    v4f f0 = *(const v4f*)(ap0 + ka);
    v4f f1 = *(const v4f*)(ap0 + ka + 4);
    v4f f2 = *(const v4f*)(ap0 + ka + 16);
    v4f f3 = *(const v4f*)(ap0 + ka + 20);
    v16bf a;
#pragma unroll
    for (int i = 0; i < 4; ++i) {
      a[i]      = (__bf16)f0[i];
      a[i + 4]  = (__bf16)f1[i];
      a[i + 8]  = (__bf16)f2[i];
      a[i + 12] = (__bf16)f3[i];
    }

    const __bf16(*bt)[40] = sB[kc & 1];
#pragma unroll
    for (int ct = 0; ct < 4; ++ct) {
      const __bf16* bp = &bt[ct * 16 + l16][hi * 8];
      v16bf b = cat16(*(const v8bf*)bp, *(const v8bf*)(bp + 16));
      acc[ct] = wmma_bf16(a, b, acc[ct]);
    }
    __syncthreads();                 // done reading before buffer reuse
  }

#pragma unroll
  for (int ct = 0; ct < 4; ++ct) {
    const int n = n0 + ct * 16 + l16;
#pragma unroll
    for (int r = 0; r < 8; ++r) {
      const int grow = row0 + r + 8 * hi;
      const __bf16 val = (__bf16)acc[ct][r];
      if (TRANSPOSE) {
        const int b = grow >> 11;
        const int s = grow & (SS - 1);
        Out[((size_t)b * EE + n) * SS + s] = val;
      } else {
        Out[(size_t)grow * EE + n] = val;
      }
    }
  }
}

// ---------------------------------------------------------------------------
// Flash-style attention. One workgroup per (batch, 16-query tile); 8 waves.
// Q tile (16x1024 bf16, reused 128x) is async-staged to LDS once.
// Per 128-key block: wave w -> 16x16 score tile over E (32 WMMAs), online
// softmax in LDS, then wave accumulates its 128 E-columns of O (32 WMMAs).
// Post-softmax 1/sqrt(E)=1/32 folded into epilogue with 1/rowsum.
// ---------------------------------------------------------------------------
__global__ __launch_bounds__(256) void attn_kernel(const __bf16* __restrict__ Q,
                                                   const __bf16* __restrict__ K,
                                                   const __bf16* __restrict__ VT,
                                                   float* __restrict__ Out) {
  __shared__ __bf16 sQ[16][1032];  // row padded: start banks stride 4 -> conflict-free
  __shared__ float  sS[16][128];
  __shared__ __bf16 sP[16][144];
  __shared__ float  sRed[16][16];
  __shared__ float  sMax[16];
  __shared__ float  sSum[16];
  __shared__ float  sScale[16];

  const int tid  = threadIdx.x;
  const int lane = tid & 31;
  const int wv   = tid >> 5;
  const int l16  = lane & 15;
  const int hi   = lane >> 4;
  const int b    = blockIdx.x >> 7;
  const int q0   = (blockIdx.x & 127) * 16;
  const int rrow = tid >> 4;
  const int cseg = tid & 15;

  if (tid < 16) { sMax[tid] = -3.0e38f; sSum[tid] = 0.0f; }

  // ---- async-stage the Q tile: 16 rows x 2048B = 2048 x 16B chunks
#pragma unroll
  for (int i = 0; i < 8; ++i) {
    const int cid = tid + i * 256;
    const int r = cid >> 7;          // 0..15
    const int c = cid & 127;         // 16B chunk in row
    async_g2l_b128((unsigned)(size_t)&sQ[r][c * 8],
                   Q + ((size_t)b * SS + q0 + r) * EE + c * 8);
  }

  v8f o[8];
#pragma unroll
  for (int ct = 0; ct < 8; ++ct) o[ct] = (v8f){0,0,0,0,0,0,0,0};

  wait_async_0();
  __syncthreads();

  for (int kb = 0; kb < SS / 128; ++kb) {
    const __bf16* krow = K + ((size_t)b * SS + kb * 128 + wv * 16 + l16) * EE;
    if (kb + 1 < SS / 128) {
      // warm L2/WGP$ for next block's K row and V columns
      __builtin_prefetch(krow + (size_t)128 * EE, 0, 3);
      __builtin_prefetch(VT + ((size_t)b * EE + wv * 128 + l16) * SS + (kb + 1) * 128, 0, 3);
    }

    // -------- scores: 16 queries x 16 keys per wave, K-loop over E
    v8f sc = (v8f){0,0,0,0,0,0,0,0};
    for (int kc = 0; kc < EE / 32; ++kc) {
      const __bf16* ap = &sQ[l16][kc * 32 + hi * 8];                 // LDS
      v16bf a = cat16(*(const v8bf*)ap, *(const v8bf*)(ap + 16));
      const __bf16* bp = krow + kc * 32 + hi * 8;                    // global
      v16bf bf = cat16(*(const v8bf*)bp, *(const v8bf*)(bp + 16));
      sc = wmma_bf16(a, bf, sc);
    }
#pragma unroll
    for (int r = 0; r < 8; ++r) sS[r + 8 * hi][wv * 16 + l16] = sc[r];
    __syncthreads();

    // -------- online softmax over this 128-key block
    float pm = sS[rrow][cseg * 8];
#pragma unroll
    for (int i = 1; i < 8; ++i) pm = fmaxf(pm, sS[rrow][cseg * 8 + i]);
    sRed[rrow][cseg] = pm;
    __syncthreads();
    if (cseg == 0) {
      float bm = sRed[rrow][0];
#pragma unroll
      for (int j = 1; j < 16; ++j) bm = fmaxf(bm, sRed[rrow][j]);
      const float mold = sMax[rrow];
      const float mnew = fmaxf(mold, bm);
      sScale[rrow] = __expf(mold - mnew);
      sMax[rrow]   = mnew;
    }
    __syncthreads();
    const float mnew = sMax[rrow];
    float psum = 0.0f;
#pragma unroll
    for (int i = 0; i < 8; ++i) {
      const float p = __expf(sS[rrow][cseg * 8 + i] - mnew);
      psum += p;
      sP[rrow][cseg * 8 + i] = (__bf16)p;
    }
    sRed[rrow][cseg] = psum;
    __syncthreads();
    if (cseg == 0) {
      float s = 0.0f;
#pragma unroll
      for (int j = 0; j < 16; ++j) s += sRed[rrow][j];
      sSum[rrow] = sSum[rrow] * sScale[rrow] + s;
    }

    // -------- rescale running O by exp(m_old - m_new)
#pragma unroll
    for (int r = 0; r < 8; ++r) {
      const float f = sScale[r + 8 * hi];
#pragma unroll
      for (int ct = 0; ct < 8; ++ct) o[ct][r] *= f;
    }

    // -------- O += P @ V ; wave owns E columns [wv*128, wv*128+128)
#pragma unroll
    for (int kk = 0; kk < 4; ++kk) {
      const __bf16* pp = &sP[l16][kk * 32 + hi * 8];
      v16bf a = cat16(*(const v8bf*)pp, *(const v8bf*)(pp + 16));
      const int keybase = kb * 128 + kk * 32 + hi * 8;
#pragma unroll
      for (int ct = 0; ct < 8; ++ct) {
        const int ecol = wv * 128 + ct * 16 + l16;
        const __bf16* bp = VT + ((size_t)b * EE + ecol) * SS + keybase;
        v16bf bf = cat16(*(const v8bf*)bp, *(const v8bf*)(bp + 16));
        o[ct] = wmma_bf16(a, bf, o[ct]);
      }
    }
    __syncthreads();
  }

  // -------- epilogue: O / (rowsum * 32)
#pragma unroll
  for (int r = 0; r < 8; ++r) {
    const int m = r + 8 * hi;
    const float inv = 1.0f / (sSum[m] * 32.0f);
    float* op = Out + ((size_t)b * SS + q0 + m) * EE + wv * 128 + l16;
#pragma unroll
    for (int ct = 0; ct < 8; ++ct) op[ct * 16] = o[ct][r] * inv;
  }
}

// ---------------------------------------------------------------------------
extern "C" void kernel_launch(void* const* d_in, const int* in_sizes, int n_in,
                              void* d_out, int out_size, void* d_ws, size_t ws_size,
                              hipStream_t stream) {
  (void)in_sizes; (void)n_in; (void)out_size; (void)ws_size;

  const float* values  = (const float*)d_in[0];
  const float* keys    = (const float*)d_in[1];
  const float* queries = (const float*)d_in[2];
  const float* Wv      = (const float*)d_in[3];
  const float* Wk      = (const float*)d_in[4];
  const float* Wq      = (const float*)d_in[5];

  char* ws = (char*)d_ws;
  const size_t szAct = (size_t)BB * SS * EE * 2;
  const size_t szW   = (size_t)EE * EE * 2;
  __bf16* vT  = (__bf16*)(ws);
  __bf16* kbf = (__bf16*)(ws + szAct);
  __bf16* qbf = (__bf16*)(ws + 2 * szAct);
  __bf16* WtV = (__bf16*)(ws + 3 * szAct);
  __bf16* WtK = (__bf16*)(ws + 3 * szAct + szW);
  __bf16* WtQ = (__bf16*)(ws + 3 * szAct + 2 * szW);

  wt_prep_kernel<<<(EE * EE) / 256, 256, 0, stream>>>(Wv, WtV);
  wt_prep_kernel<<<(EE * EE) / 256, 256, 0, stream>>>(Wk, WtK);
  wt_prep_kernel<<<(EE * EE) / 256, 256, 0, stream>>>(Wq, WtQ);

  dim3 pg(BB * SS / 128, EE / 64);
  proj_kernel<true ><<<pg, 256, 0, stream>>>(values,  WtV, vT);
  proj_kernel<false><<<pg, 256, 0, stream>>>(keys,    WtK, kbf);
  proj_kernel<false><<<pg, 256, 0, stream>>>(queries, WtQ, qbf);

  attn_kernel<<<BB * SS / 16, 256, 0, stream>>>(qbf, kbf, vT, (float*)d_out);
}
